// Block_79001628442785
// MI455X (gfx1250) — matrix operand order repaired and checked by reading, not compiled, not run on hardware
//
#include <hip/hip_runtime.h>

typedef unsigned short ushort_t;
typedef unsigned int uint_t;
typedef __attribute__((ext_vector_type(16))) __bf16 v16bf;
typedef __attribute__((ext_vector_type(8)))  float  v8f;
typedef __attribute__((ext_vector_type(4)))  uint_t su4;
typedef __attribute__((ext_vector_type(8)))  uint_t su8;

#define NWIN   1024
#define NTOK   64          // tokens per window
#define CDIM   512
#define NHEAD  16
#define DHEAD  32
#define MTOT   65536       // NWIN*NTOK = B*H*W
#define INNER  2048

__device__ __forceinline__ ushort_t f2bf(float f) {
    uint_t u = __float_as_uint(f);
    u += 0x7FFFu + ((u >> 16) & 1u);       // round-to-nearest-even
    return (ushort_t)(u >> 16);
}

// flat->LDS: hardware uses addr[31:0] as the LDS byte address (ISA 10.2)
__device__ __forceinline__ uint_t lds_off(const void* p) {
    return (uint_t)(unsigned long long)(uintptr_t)p;
}

// A fragment (16x32, bf16).  Lane l: row m=l&15, khalf=l>>4.
// Elements 0..7  -> k = khalf*8 + 0..7       (16B contiguous)
// Elements 8..15 -> k = 16 + khalf*8 + 0..7  (16B contiguous)
__device__ __forceinline__ v16bf load_frag_a(const ushort_t* base, int lda) {
    int l  = threadIdx.x & 31;
    int m  = l & 15;
    int kh = l >> 4;
    const ushort_t* p = base + m * lda + kh * 8;
    union { v16bf v; uint4 u[2]; } r;
    r.u[0] = *(const uint4*)(p);
    r.u[1] = *(const uint4*)(p + 16);
    return r.v;
}

// B fragment (32x16, bf16) from LDS stored as Bt[N][K] (row = n, contiguous k).
// Lane l: col n=l&15, khalf=l>>4; elements 0..7 -> k = khalf*8 + 0..7, 8..15 -> k = 16 + khalf*8 + 0..7
__device__ __forceinline__ v16bf load_frag_b(const ushort_t* base, int ldb) {
    int l  = threadIdx.x & 31;
    int n  = l & 15;
    int kh = l >> 4;
    const ushort_t* p = base + n * ldb + kh * 8;
    union { v16bf v; uint4 u[2]; } r;
    r.u[0] = *(const uint4*)(p);
    r.u[1] = *(const uint4*)(p + 16);
    return r.v;
}

__device__ __forceinline__ v8f wmma_bf16(v16bf a, v16bf b, v8f c) {
    return __builtin_amdgcn_wmma_f32_16x16x32_bf16(
        false, a, false, b, (short)0, c, false, false);
}

// TDM: DMA a [rows x 64] bf16 tile (row stride K elements) from global to LDS.
// D# layout per CDNA5 ISA ch.8: group0 = {flags, lds_addr, global_addr, type=2},
// group1 = {data_size=2B, tensor_dim0=64, tensor_dim1=rows, tile_dim0=64,
//           tile_dim1=rows, tensor_dim0_stride=K}
__device__ __forceinline__ void tdm_load_tile_b64xR(
    const ushort_t* gsrc, uint_t lds_addr, uint_t rows, uint_t strideK)
{
    unsigned long long ga = (unsigned long long)(uintptr_t)gsrc;
    su4 g0;
    g0[0] = 1u;                                   // count=1, user mode
    g0[1] = lds_addr;                             // lds_addr [63:32]
    g0[2] = (uint_t)ga;                           // global_addr lo
    g0[3] = (uint_t)(ga >> 32) | (2u << 30);      // global_addr hi | type=2
    su8 g1;
    g1[0] = 0x10000u;                             // data_size=1 (2 bytes)
    g1[1] = (64u << 16);                          // tensor_dim0[15:0]=64 @ [63:48]
    g1[2] = (rows << 16);                         // tensor_dim1[15:0]=rows @ [95:80]
    g1[3] = (64u << 16);                          // tile_dim0=64 @ [127:112]
    g1[4] = rows;                                 // tile_dim1=rows @ [143:128]
    g1[5] = strideK;                              // tensor_dim0_stride[31:0]
    g1[6] = 0u;                                   // stride hi / dim1_stride lo
    g1[7] = 0u;
    asm volatile("tensor_load_to_lds %0, %1" :: "s"(g0), "s"(g1) : "memory");
}

// ---------------------------------------------------------------------------
// Kernel 1: windowed attention, one block per (window, head)
//   attn = softmax(Q Kt * scale) + bias(after softmax);  out = attn @ V
// ---------------------------------------------------------------------------
__global__ __launch_bounds__(256) void attn_kernel(
    const float* __restrict__ q, const float* __restrict__ k,
    const float* __restrict__ v, const float* __restrict__ rpb,
    ushort_t* __restrict__ ao)
{
    __shared__ __align__(16) ushort_t Qb[NTOK * DHEAD];   // [64][32]
    __shared__ __align__(16) ushort_t Kb[NTOK * DHEAD];   // [64][32]
    __shared__ __align__(16) ushort_t Vt[DHEAD * NTOK];   // [32][64] (transposed)
    __shared__ __align__(16) float    Sf[NTOK * NTOK];    // scores f32
    __shared__ __align__(16) ushort_t Ab[NTOK * NTOK];    // attn bf16

    const int blk  = blockIdx.x;
    const int w    = blk >> 4;
    const int h    = blk & 15;
    const int tid  = threadIdx.x;
    const int wid  = tid >> 5;
    const int lane = tid & 31;

    const size_t base = (size_t)w * NTOK * CDIM + h * DHEAD;
    const float* qp = q + base;
    const float* kp = k + base;
    const float* vp = v + base;

    // ---- load & convert: each thread 8 consecutive f32 ----
    {
        int n = tid >> 2;
        int d = (tid & 3) * 8;
        float4 q0 = *(const float4*)(qp + n * CDIM + d);
        float4 q1 = *(const float4*)(qp + n * CDIM + d + 4);
        float4 k0 = *(const float4*)(kp + n * CDIM + d);
        float4 k1 = *(const float4*)(kp + n * CDIM + d + 4);
        float4 v0 = *(const float4*)(vp + n * CDIM + d);
        float4 v1 = *(const float4*)(vp + n * CDIM + d + 4);
        uint_t* qw = (uint_t*)&Qb[n * DHEAD + d];
        qw[0] = f2bf(q0.x) | ((uint_t)f2bf(q0.y) << 16);
        qw[1] = f2bf(q0.z) | ((uint_t)f2bf(q0.w) << 16);
        qw[2] = f2bf(q1.x) | ((uint_t)f2bf(q1.y) << 16);
        qw[3] = f2bf(q1.z) | ((uint_t)f2bf(q1.w) << 16);
        uint_t* kw = (uint_t*)&Kb[n * DHEAD + d];
        kw[0] = f2bf(k0.x) | ((uint_t)f2bf(k0.y) << 16);
        kw[1] = f2bf(k0.z) | ((uint_t)f2bf(k0.w) << 16);
        kw[2] = f2bf(k1.x) | ((uint_t)f2bf(k1.y) << 16);
        kw[3] = f2bf(k1.z) | ((uint_t)f2bf(k1.w) << 16);
        float vv[8] = {v0.x, v0.y, v0.z, v0.w, v1.x, v1.y, v1.z, v1.w};
#pragma unroll
        for (int i = 0; i < 8; ++i) Vt[(d + i) * NTOK + n] = f2bf(vv[i]);
    }
    __syncthreads();

    // ---- phase 1: S = Q @ K^T * scale  (16 tiles, 2 per wave) ----
    const float scale = 0.17677669529663687f;   // 1/sqrt(32)
#pragma unroll
    for (int t = 0; t < 2; ++t) {
        int ti = wid * 2 + t;
        int tm = ti >> 2, tn = ti & 3;
        v16bf a = load_frag_a(Qb + tm * 16 * DHEAD, DHEAD);
        v16bf b = load_frag_b(Kb + tn * 16 * DHEAD, DHEAD);
        v8f c = {};
        c = wmma_bf16(a, b, c);
        int mrow = tm * 16 + ((lane >> 4) << 3);
        int ncol = tn * 16 + (lane & 15);
#pragma unroll
        for (int r = 0; r < 8; ++r) Sf[(mrow + r) * NTOK + ncol] = c[r] * scale;
    }
    __syncthreads();

    // ---- phase 2: softmax over rows, then add rel-pos bias, cast bf16 ----
    if (tid < NTOK) {
        const int i  = tid;
        const int yi = i >> 3, xi = i & 7;
        float mx = -3.4e38f;
        for (int j = 0; j < NTOK; ++j) mx = fmaxf(mx, Sf[i * NTOK + j]);
        float sum = 0.f;
        for (int j = 0; j < NTOK; ++j) sum += __expf(Sf[i * NTOK + j] - mx);
        float inv = 1.f / sum;
        for (int j = 0; j < NTOK; ++j) {
            int yj = j >> 3, xj = j & 7;
            int idx = (yi - yj + 7) * 15 + (xi - xj + 7);
            float p = __expf(Sf[i * NTOK + j] - mx) * inv + rpb[idx * NHEAD + h];
            Ab[i * NTOK + j] = f2bf(p);
        }
    }
    __syncthreads();

    // ---- phase 3: out = attn @ V  (8 tiles, 1 per wave; K=64 -> 2 wmma) ----
    {
        int tm = wid >> 1, tn = wid & 1;
        v8f acc = {};
#pragma unroll
        for (int kk = 0; kk < 64; kk += 32) {
            v16bf a = load_frag_a(Ab + tm * 16 * NTOK + kk, NTOK);
            v16bf b = load_frag_b(Vt + tn * 16 * NTOK + kk, NTOK);
            acc = wmma_bf16(a, b, acc);
        }
        int mrow = tm * 16 + ((lane >> 4) << 3);
        int ncol = tn * 16 + (lane & 15);
        ushort_t* aop = ao + (size_t)w * NTOK * CDIM + h * DHEAD;
#pragma unroll
        for (int r = 0; r < 8; ++r)
            aop[(size_t)(mrow + r) * CDIM + ncol] = f2bf(acc[r]);
    }
}

// ---------------------------------------------------------------------------
// Kernel 2/4/5: GEMM  out = epilogue(A[M,K](bf16) @ W[N,K]^T + bias)
// Block tile 128x64, BK=64; A tile comes in via the Tensor Data Mover.
// 8 waves in 4x2 layout, each owns 2x2 wmma tiles -> 8 v_wmma per K-step.
// MODE 0: proj + window-reverse + x + gamma1*(...)        -> f32 d_out
// MODE 1: gelu(exact)                                     -> bf16 ws
// MODE 2: x1 + gamma2*(...)                               -> f32 d_out
// ---------------------------------------------------------------------------
template <int MODE>
__global__ __launch_bounds__(256) void gemm_kernel(
    const ushort_t* __restrict__ A, const float* __restrict__ W,
    const float* __restrict__ bias, const float* __restrict__ resid,
    const float* __restrict__ gamma, float* __restrict__ out_f32,
    ushort_t* __restrict__ out_bf16, int K, int N)
{
    __shared__ __align__(16) ushort_t As[128 * 64];   // 16 KB
    __shared__ __align__(16) ushort_t Bs[64 * 64];    //  8 KB

    const int m0   = blockIdx.x * 128;
    const int n0   = blockIdx.y * 64;
    const int tid  = threadIdx.x;
    const int wid  = tid >> 5;
    const int lane = tid & 31;
    const int wm   = wid >> 1;      // 0..3
    const int wn   = wid & 1;       // 0..1

    v8f acc[2][2] = {};

    for (int k0 = 0; k0 < K; k0 += 64) {
        __syncthreads();            // LDS free for reuse
        if (wid == 0) {
            // one TDM op loads the whole 128x64 bf16 A tile into LDS
            tdm_load_tile_b64xR(A + (size_t)m0 * K + k0, lds_off(&As[0]),
                                128u, (uint_t)K);
        }
        // W tile: 64x64 f32 -> bf16, stored [n][k]
#pragma unroll
        for (int it = 0; it < 4; ++it) {
            int c   = tid + it * 256;
            int row = c >> 4;
            int kc  = (c & 15) * 4;
            float4 f = *(const float4*)(W + (size_t)(n0 + row) * K + k0 + kc);
            uint2 u;
            u.x = f2bf(f.x) | ((uint_t)f2bf(f.y) << 16);
            u.y = f2bf(f.z) | ((uint_t)f2bf(f.w) << 16);
            *(uint2*)&Bs[row * 64 + kc] = u;
        }
        __builtin_amdgcn_s_wait_tensorcnt(0);
        __syncthreads();

#pragma unroll
        for (int kk = 0; kk < 64; kk += 32) {
            v16bf a0 = load_frag_a(As + (wm * 2 + 0) * 16 * 64 + kk, 64);
            v16bf a1 = load_frag_a(As + (wm * 2 + 1) * 16 * 64 + kk, 64);
            v16bf b0 = load_frag_b(Bs + (wn * 2 + 0) * 16 * 64 + kk, 64);
            v16bf b1 = load_frag_b(Bs + (wn * 2 + 1) * 16 * 64 + kk, 64);
            acc[0][0] = wmma_bf16(a0, b0, acc[0][0]);
            acc[0][1] = wmma_bf16(a0, b1, acc[0][1]);
            acc[1][0] = wmma_bf16(a1, b0, acc[1][0]);
            acc[1][1] = wmma_bf16(a1, b1, acc[1][1]);
        }
    }

    // epilogue
#pragma unroll
    for (int i = 0; i < 2; ++i) {
#pragma unroll
        for (int j = 0; j < 2; ++j) {
            int tile_m = wm * 2 + i;
            int tile_n = wn * 2 + j;
            int mbase  = tile_m * 16 + ((lane >> 4) << 3);
            int gn     = n0 + tile_n * 16 + (lane & 15);
            float bn = bias[gn];
#pragma unroll
            for (int r = 0; r < 8; ++r) {
                int gm  = m0 + mbase + r;
                float val = acc[i][j][r] + bn;
                if (MODE == 0) {
                    // window-reverse: row gm in (win, tok) order -> spatial
                    int wi = gm >> 6, n = gm & 63;
                    int b  = wi >> 6, wy = (wi >> 3) & 7, wx = wi & 7;
                    int iy = n >> 3,  ix = n & 7;
                    size_t s = ((size_t)b * 64 + wy * 8 + iy) * 64 + wx * 8 + ix;
                    out_f32[s * CDIM + gn] =
                        resid[s * CDIM + gn] + gamma[gn] * val;
                } else if (MODE == 1) {
                    float g = 0.5f * val * (1.f + erff(val * 0.70710678118654752f));
                    out_bf16[(size_t)gm * N + gn] = f2bf(g);
                } else {
                    size_t s = (size_t)gm * CDIM + gn;
                    out_f32[s] = resid[s] + gamma[gn] * val;
                }
            }
        }
    }
}

// ---------------------------------------------------------------------------
// Kernel 3: LayerNorm over C=512, one block per token row -> bf16
// ---------------------------------------------------------------------------
__global__ __launch_bounds__(256) void ln_kernel(
    const float* __restrict__ x1, const float* __restrict__ w,
    const float* __restrict__ b, ushort_t* __restrict__ xn)
{
    __shared__ float red[256];
    const int tid = threadIdx.x;
    const size_t row = blockIdx.x;
    const float* p = x1 + row * CDIM;
    float v0 = p[tid], v1 = p[tid + 256];

    red[tid] = v0 + v1;
    __syncthreads();
    for (int st = 128; st > 0; st >>= 1) {
        if (tid < st) red[tid] += red[tid + st];
        __syncthreads();
    }
    float mu = red[0] * (1.f / CDIM);
    __syncthreads();

    float d0 = v0 - mu, d1 = v1 - mu;
    red[tid] = d0 * d0 + d1 * d1;
    __syncthreads();
    for (int st = 128; st > 0; st >>= 1) {
        if (tid < st) red[tid] += red[tid + st];
        __syncthreads();
    }
    float rs = rsqrtf(red[0] * (1.f / CDIM) + 1e-5f);

    xn[row * CDIM + tid]       = f2bf(d0 * rs * w[tid] + b[tid]);
    xn[row * CDIM + tid + 256] = f2bf(d1 * rs * w[tid + 256] + b[tid + 256]);
}

// ---------------------------------------------------------------------------
extern "C" void kernel_launch(void* const* d_in, const int* in_sizes, int n_in,
                              void* d_out, int out_size, void* d_ws, size_t ws_size,
                              hipStream_t stream) {
    (void)in_sizes; (void)n_in; (void)out_size; (void)ws_size;
    const float* x      = (const float*)d_in[0];
    const float* q      = (const float*)d_in[1];
    const float* k      = (const float*)d_in[2];
    const float* v      = (const float*)d_in[3];
    const float* proj_w = (const float*)d_in[4];
    const float* proj_b = (const float*)d_in[5];
    const float* rpb    = (const float*)d_in[6];
    const float* gamma1 = (const float*)d_in[7];
    const float* ln2_w  = (const float*)d_in[8];
    const float* ln2_b  = (const float*)d_in[9];
    const float* fc1_w  = (const float*)d_in[10];
    const float* fc1_b  = (const float*)d_in[11];
    const float* fc2_w  = (const float*)d_in[12];
    const float* fc2_b  = (const float*)d_in[13];
    const float* gamma2 = (const float*)d_in[14];
    float* out = (float*)d_out;

    // ws layout: [0, 32Mi ushort) = ao, later reused as xn;  [32Mi, ...) = h1
    ushort_t* ao = (ushort_t*)d_ws;                                   // 65536*512 bf16
    ushort_t* xn = (ushort_t*)d_ws;                                   // reuses ao region
    ushort_t* h1 = (ushort_t*)((char*)d_ws + (size_t)MTOT * CDIM * 2); // 65536*2048 bf16

    attn_kernel<<<NWIN * NHEAD, 256, 0, stream>>>(q, k, v, rpb, ao);

    gemm_kernel<0><<<dim3(MTOT / 128, CDIM / 64), 256, 0, stream>>>(
        ao, proj_w, proj_b, x, gamma1, out, nullptr, CDIM, CDIM);

    ln_kernel<<<MTOT, 256, 0, stream>>>(out, ln2_w, ln2_b, xn);

    gemm_kernel<1><<<dim3(MTOT / 128, INNER / 64), 256, 0, stream>>>(
        xn, fc1_w, fc1_b, nullptr, nullptr, nullptr, h1, CDIM, INNER);

    gemm_kernel<2><<<dim3(MTOT / 128, CDIM / 64), 256, 0, stream>>>(
        h1, fc2_w, fc2_b, out, gamma2, out, nullptr, INNER, CDIM);
}
